// MultiHeadAttention_22308060135993
// MI455X (gfx1250) — matrix operand order repaired and checked by reading, not compiled, not run on hardware
//
#include <hip/hip_runtime.h>

// ---------------------------------------------------------------------------
// MI455X (gfx1250) multi-head attention, bf16 WMMA path, wave32.
//   B=4, S=2048, E=1024, H=16, D=64
// Pipeline:
//   1) cvt  : fp32 -> bf16 activations (key/query/value)
//   2) trn  : fp32 -> bf16 weight transpose  Wt[n][k] = W[k][n]
//   3) gemm : X @ Wt -> Q,K in (b,h,s,d) bf16 ; V in (b,h,d,s) bf16
//   4) attn : causal flash attention, 4 q-tiles/block sharing async-staged
//             K/V LDS tiles (global_load_async_to_lds_b128, ASYNCcnt)
//   5) gemm : attn @ Wot -> f32 output (b,s,e)
// ---------------------------------------------------------------------------

typedef __bf16 bf16_t;
typedef __attribute__((ext_vector_type(16))) __bf16 v16bf;
typedef __attribute__((ext_vector_type(8)))  __bf16 v8bf;
typedef __attribute__((ext_vector_type(8)))  float  v8f;

#define B_ 4
#define S_ 2048
#define E_ 1024
#define H_ 16
#define D_ 64

#define KT_STRIDE 72   // LDS K-tile row stride (elems): 32 rows x 64 d, padded
#define VT_STRIDE 40   // LDS V-tile row stride (elems): 64 rows x 32 keys, padded

static __device__ __forceinline__ v16bf frag_combine(v8bf lo, v8bf hi) {
  v16bf r;
#pragma unroll
  for (int i = 0; i < 8; ++i) { r[i] = lo[i]; r[i + 8] = hi[i]; }
  return r;
}

// A fragment, 16x32 (MxK), bf16, row-major source with row stride ld.
// ISA 16-bit A layout: lane m = lane&15, g = lane>>4; per lane the 16 values
// are k = [g*8, g*8+8) then k = [16+g*8, 16+g*8+8)  -> two b128 loads.
static __device__ __forceinline__ v16bf load_a(const bf16_t* base, int ld, int lane) {
  int m = lane & 15, g = lane >> 4;
  const bf16_t* row = base + m * ld;
  v8bf lo = *(const v8bf*)(row + g * 8);
  v8bf hi = *(const v8bf*)(row + 16 + g * 8);
  return frag_combine(lo, hi);
}

// B fragment, 32x16 (KxN), bf16. Source must be "column-contiguous in k":
// element (k,n) at base[n*ldc + k]. Lane n = lane&15, k = g*8+i then 16+g*8+i.
static __device__ __forceinline__ v16bf load_b(const bf16_t* base, int ldc, int lane) {
  int n = lane & 15, g = lane >> 4;
  const bf16_t* col = base + n * ldc + g * 8;
  v8bf lo = *(const v8bf*)(col);
  v8bf hi = *(const v8bf*)(col + 16);
  return frag_combine(lo, hi);
}

static __device__ __forceinline__ v8f wmma_bf16(v16bf a, v16bf b, v8f c) {
  return __builtin_amdgcn_wmma_f32_16x16x32_bf16(false, a, false, b, (short)0, c,
                                                 false, false);
}

// Async copy of one 16-byte chunk per lane into LDS (CDNA5 ASYNCcnt path).
static __device__ __forceinline__ void async_b128(unsigned lds_off, const void* g) {
  asm volatile("global_load_async_to_lds_b128 %0, %1, off"
               :: "v"(lds_off), "v"(g) : "memory");
}

// ------------------------------- prep kernels -------------------------------

__global__ __launch_bounds__(256) void mha_cvt_f32_bf16(
    const float* __restrict__ in, bf16_t* __restrict__ out, int n) {
  int i = blockIdx.x * blockDim.x + threadIdx.x;
  int stride = gridDim.x * blockDim.x;
  for (; i < n; i += stride) out[i] = (bf16_t)in[i];
}

// Wt[n][k] = (bf16) W[k][n], W is E_ x E_ fp32 row-major.
__global__ __launch_bounds__(256) void mha_transpose_cvt(
    const float* __restrict__ Win, bf16_t* __restrict__ Wt) {
  __shared__ float tile[32][33];
  int tx = threadIdx.x, ty = threadIdx.y;
  int x = blockIdx.x * 32 + tx;   // n
  int y0 = blockIdx.y * 32;       // k base
#pragma unroll
  for (int j = ty; j < 32; j += 8)
    tile[j][tx] = Win[(y0 + j) * E_ + x];
  __syncthreads();
  int xx = y0 + tx;               // k
  int yy = blockIdx.x * 32;       // n base
#pragma unroll
  for (int j = ty; j < 32; j += 8)
    Wt[(yy + j) * E_ + xx] = (bf16_t)tile[tx][j];
}

// -------------------------------- GEMM --------------------------------------
// C(8192x1024) = X(8192x1024,bf16,row-major) @ Wt^T  (Wt is [n][k] bf16)
// One wave -> 16x64 tile. Epilogue modes:
//   0: bf16 -> (b,h,s,d)      1: bf16 -> (b,h,d,s)      2: f32 -> row-major
__global__ __launch_bounds__(128) void mha_gemm(
    const bf16_t* __restrict__ X, const bf16_t* __restrict__ Wt,
    bf16_t* __restrict__ outb, float* __restrict__ outf, int mode) {
  int lane = threadIdx.x;
  int rt = blockIdx.x;                        // 0..511
  int ct = blockIdx.y * 4 + threadIdx.y;      // 0..15
  int row0 = rt * 16;
  int col0 = ct * 64;

  v8f acc[4] = {};
  const bf16_t* xrow = X + (size_t)row0 * E_;
  for (int kk = 0; kk < E_; kk += 32) {
    v16bf a = load_a(xrow + kk, E_, lane);
#pragma unroll
    for (int t = 0; t < 4; ++t) {
      v16bf b = load_b(Wt + (size_t)(col0 + t * 16) * E_ + kk, E_, lane);
      acc[t] = wmma_bf16(a, b, acc[t]);
    }
  }

  int n = lane & 15, g = lane >> 4;
#pragma unroll
  for (int t = 0; t < 4; ++t) {
    int col = col0 + t * 16 + n;
#pragma unroll
    for (int r = 0; r < 8; ++r) {
      int row = row0 + r + 8 * g;             // C layout: m = r + 8*g, n = lane&15
      float v = acc[t][r];
      if (mode == 2) {
        outf[(size_t)row * E_ + col] = v;
      } else {
        int b = row >> 11, s = row & (S_ - 1);
        int h = col >> 6,  d = col & 63;
        if (mode == 0)
          outb[(((size_t)(b * H_ + h)) * S_ + s) * D_ + d] = (bf16_t)v;
        else
          outb[(((size_t)(b * H_ + h)) * D_ + d) * S_ + s] = (bf16_t)v;
      }
    }
  }
}

// ------------------------------ attention -----------------------------------
// Block = 4 waves = 4 consecutive 16-query tiles of one (b,h). Key tiles of 32
// are staged once per block into double-buffered LDS via async copies and
// consumed by all 4 waves. Online softmax per wave.
// Q,K in (b,h,s,d) bf16 ; V in (b,h,d,s) bf16 ; out A in (b,s,h*64+d) bf16.

// Stage one key tile: K tile = 32 rows x 128B contiguous in global;
// V tile = 64 rows (d) x 64B, row stride S_*2 bytes. 4 async instr per wave.
static __device__ __forceinline__ void attn_stage(
    const bf16_t* Kg, const bf16_t* Vg, bf16_t* sKbuf, bf16_t* sVbuf, int tid) {
#pragma unroll
  for (int j = 0; j < 2; ++j) {
    int c = tid + j * 128;                    // 256 chunks of 16B
    unsigned l = (unsigned)(size_t)sKbuf + (unsigned)((c >> 3) * (KT_STRIDE * 2) + (c & 7) * 16);
    async_b128(l, (const char*)Kg + c * 16);
  }
#pragma unroll
  for (int j = 0; j < 2; ++j) {
    int c = tid + j * 128;                    // 256 chunks of 16B
    unsigned l = (unsigned)(size_t)sVbuf + (unsigned)((c >> 2) * (VT_STRIDE * 2) + (c & 3) * 16);
    async_b128(l, (const char*)Vg + (c >> 2) * (S_ * 2) + (c & 3) * 16);
  }
}

__global__ __launch_bounds__(128) void mha_attn(
    const bf16_t* __restrict__ Q, const bf16_t* __restrict__ K,
    const bf16_t* __restrict__ V, bf16_t* __restrict__ A) {
  __shared__ __align__(16) bf16_t sK[2][32 * KT_STRIDE];
  __shared__ __align__(16) bf16_t sV[2][64 * VT_STRIDE];
  __shared__ __align__(16) bf16_t sP[4][16 * 40];

  int lane = threadIdx.x;
  int w = threadIdx.y;
  int tid = w * 32 + lane;
  bf16_t* P = &sP[w][0];

  int idx = blockIdx.x * 4 + w;               // 0..8191
  int bh = idx >> 7;                          // uniform across block
  int qt = idx & 127;
  int q0 = qt * 16;
  int qt_hi = (blockIdx.x * 4 + 3) & 127;     // highest q-tile in block
  int nkt = (qt_hi * 16 + 16 + 31) / 32;      // uniform trip count
  int nLane = lane & 15, g = lane >> 4;

  const bf16_t* qbase = Q + ((size_t)bh * S_ + q0) * D_;
  v16bf qa0 = load_a(qbase, D_, lane);        // d = 0..31
  v16bf qa1 = load_a(qbase + 32, D_, lane);   // d = 32..63

  float Mr[8], Lr[8];
  v8f acc[4] = {};
#pragma unroll
  for (int r = 0; r < 8; ++r) { Mr[r] = -3.0e38f; Lr[r] = 0.0f; }

  const bf16_t* Kbh = K + (size_t)bh * S_ * D_;
  const bf16_t* Vbh = V + (size_t)bh * D_ * S_;

  attn_stage(Kbh, Vbh, sK[0], sV[0], tid);

  for (int kt = 0; kt < nkt; ++kt) {
    int buf = kt & 1;
    int k0 = kt * 32;
    if (kt + 1 < nkt) {
      attn_stage(Kbh + (size_t)(k0 + 32) * D_, Vbh + (k0 + 32),
                 sK[buf ^ 1], sV[buf ^ 1], tid);
      // 8 outstanding, in-order: <=4 retires exactly the current buffer.
      asm volatile("s_wait_asynccnt 0x4" ::: "memory");
    } else {
      asm volatile("s_wait_asynccnt 0x0" ::: "memory");
    }
    __syncthreads();                          // staged tile visible to all waves

    bool active = (k0 <= q0 + 15);            // causal: this wave needs the tile
    if (active) {
      const bf16_t* kl = &sK[buf][0];
      const bf16_t* vl = &sV[buf][0];
      // scores: 16 queries x 32 keys, K-dim = D = 64
      v8f c0 = {}, c1 = {};
      c0 = wmma_bf16(qa0, load_b(kl, KT_STRIDE, lane), c0);
      c0 = wmma_bf16(qa1, load_b(kl + 32, KT_STRIDE, lane), c0);
      c1 = wmma_bf16(qa0, load_b(kl + 16 * KT_STRIDE, KT_STRIDE, lane), c1);
      c1 = wmma_bf16(qa1, load_b(kl + 16 * KT_STRIDE + 32, KT_STRIDE, lane), c1);

      bool maskTile = (k0 + 31 > q0);
#pragma unroll
      for (int r = 0; r < 8; ++r) {
        int qrow = q0 + r + 8 * g;
        float s0 = (float)c0[r] * 0.125f;     // 1/sqrt(64)
        float s1 = (float)c1[r] * 0.125f;
        if (maskTile) {
          if (k0 + nLane      > qrow) s0 = -3.0e38f;
          if (k0 + 16 + nLane > qrow) s1 = -3.0e38f;
        }
        float mx = fmaxf(s0, s1);
#pragma unroll
        for (int off = 1; off < 16; off <<= 1) mx = fmaxf(mx, __shfl_xor(mx, off, 32));
        float mnew = fmaxf(Mr[r], mx);
        float alpha = __expf(Mr[r] - mnew);
        float p0 = __expf(s0 - mnew);
        float p1 = __expf(s1 - mnew);
        float ls = p0 + p1;
#pragma unroll
        for (int off = 1; off < 16; off <<= 1) ls += __shfl_xor(ls, off, 32);
        Lr[r] = Lr[r] * alpha + ls;
        Mr[r] = mnew;
#pragma unroll
        for (int t = 0; t < 4; ++t) acc[t][r] *= alpha;
        P[(r + 8 * g) * 40 + nLane]      = (bf16_t)p0;   // C layout -> LDS
        P[(r + 8 * g) * 40 + 16 + nLane] = (bf16_t)p1;
      }
      // Wave-private LDS round trip: same-wave DS ops are in order; the
      // wave_barrier pins compiler scheduling across the store->load edge.
      __builtin_amdgcn_wave_barrier();
      v16bf pa;
      {
        const bf16_t* prow = P + (lane & 15) * 40;
        v8bf lo = *(const v8bf*)(prow + g * 8);
        v8bf hi = *(const v8bf*)(prow + 16 + g * 8);
        pa = frag_combine(lo, hi);
      }
      __builtin_amdgcn_wave_barrier();

      // acc += P(16x32) @ V(32x64): V tile stored (d, key) -> direct B frags.
#pragma unroll
      for (int t = 0; t < 4; ++t) {
        v16bf bv = load_b(vl + t * 16 * VT_STRIDE, VT_STRIDE, lane);
        acc[t] = wmma_bf16(pa, bv, acc[t]);
      }
    }
    __syncthreads();                          // done reading before re-stage
  }

  int b = bh >> 4, h = bh & 15;
#pragma unroll
  for (int t = 0; t < 4; ++t) {
#pragma unroll
    for (int r = 0; r < 8; ++r) {
      int qrow = q0 + r + 8 * g;
      int d = t * 16 + nLane;
      float v = acc[t][r] / Lr[r];
      A[((size_t)(b * S_ + qrow)) * E_ + h * D_ + d] = (bf16_t)v;
    }
  }
}

// ------------------------------ launcher ------------------------------------

extern "C" void kernel_launch(void* const* d_in, const int* in_sizes, int n_in,
                              void* d_out, int out_size, void* d_ws, size_t ws_size,
                              hipStream_t stream) {
  (void)in_sizes; (void)n_in; (void)out_size; (void)ws_size;
  const float* key   = (const float*)d_in[0];
  const float* query = (const float*)d_in[1];
  const float* value = (const float*)d_in[2];
  const float* in_w  = (const float*)d_in[3];   // (3, E, E): [0]=K, [1]=Q, [2]=V
  const float* out_w = (const float*)d_in[4];   // (E, E)
  float* out = (float*)d_out;

  const size_t ACT = (size_t)B_ * S_ * E_;      // 8388608
  const size_t WEL = (size_t)E_ * E_;           // 1048576
  bf16_t* ws  = (bf16_t*)d_ws;
  bf16_t* Xk  = ws;
  bf16_t* Xq  = Xk + ACT;
  bf16_t* Xv  = Xq + ACT;
  bf16_t* Wkt = Xv + ACT;
  bf16_t* Wqt = Wkt + WEL;
  bf16_t* Wvt = Wqt + WEL;
  bf16_t* Wot = Wvt + WEL;
  bf16_t* Qb  = Wot + WEL;                      // (b,h,s,d)
  bf16_t* Kb  = Qb + ACT;                       // (b,h,s,d)
  bf16_t* Vb  = Kb + ACT;                       // (b,h,d,s)
  bf16_t* Ab  = Vb + ACT;                       // (b,s,e)

  // 1) activation converts
  mha_cvt_f32_bf16<<<dim3(1024), dim3(256), 0, stream>>>(key,   Xk, (int)ACT);
  mha_cvt_f32_bf16<<<dim3(1024), dim3(256), 0, stream>>>(query, Xq, (int)ACT);
  mha_cvt_f32_bf16<<<dim3(1024), dim3(256), 0, stream>>>(value, Xv, (int)ACT);

  // 2) weight transposes
  mha_transpose_cvt<<<dim3(32, 32), dim3(32, 8), 0, stream>>>(in_w + 0 * WEL, Wkt);
  mha_transpose_cvt<<<dim3(32, 32), dim3(32, 8), 0, stream>>>(in_w + 1 * WEL, Wqt);
  mha_transpose_cvt<<<dim3(32, 32), dim3(32, 8), 0, stream>>>(in_w + 2 * WEL, Wvt);
  mha_transpose_cvt<<<dim3(32, 32), dim3(32, 8), 0, stream>>>(out_w, Wot);

  // 3) projections
  dim3 gb(32, 4), gg(512, 4);
  mha_gemm<<<gg, gb, 0, stream>>>(Xk, Wkt, Kb, nullptr, 0);
  mha_gemm<<<gg, gb, 0, stream>>>(Xq, Wqt, Qb, nullptr, 0);
  mha_gemm<<<gg, gb, 0, stream>>>(Xv, Wvt, Vb, nullptr, 1);

  // 4) causal attention (async K/V staging shared by 4 waves/block)
  mha_attn<<<dim3(2048), dim3(32, 4), 0, stream>>>(Qb, Kb, Vb, Ab);

  // 5) output projection (f32 epilogue)
  mha_gemm<<<gg, gb, 0, stream>>>(Ab, Wot, nullptr, out, 2);
}
